// SingleHeadAttention_51453708206151
// MI455X (gfx1250) — matrix operand order repaired and checked by reading, not compiled, not run on hardware
//
#include <hip/hip_runtime.h>
#include <hip/hip_bf16.h>

typedef __attribute__((ext_vector_type(16))) _Float16 v16h;
typedef __attribute__((ext_vector_type(8)))  float    v8f;

#define DMODEL 1024
#define DK     128
#define BATCH  4
#define SEQ    4096
#define BS     (BATCH * SEQ)

union V16H {
    v16h      h;
    uint4     u[2];
    _Float16  e[16];
};

__device__ __forceinline__ v8f wmma_f16(v16h a, v16h b, v8f c) {
    return __builtin_amdgcn_wmma_f32_16x16x32_f16(
        /*neg_a=*/false, a, /*neg_b=*/false, b,
        /*c_mod=*/(short)0, c, /*reuse_a=*/false, /*reuse_b=*/false);
}

// ---------------------------------------------------------------------------
// Stage 1: weight transpose + f32->f16.  W[1024][128] -> WT[128][1024] (f16)
// ---------------------------------------------------------------------------
__global__ void wprep_kernel(const float* __restrict__ W, _Float16* __restrict__ WT) {
    int t = blockIdx.x * blockDim.x + threadIdx.x;
    if (t >= DMODEL * DK) return;
    int k = t >> 7;        // 0..1023
    int n = t & 127;       // 0..127
    WT[n * DMODEL + k] = (_Float16)W[k * DK + n];
}

// ---------------------------------------------------------------------------
// Stage 2: projection GEMM  Out = X[BS][1024] @ W[1024][128] + bias
//   MODE 0: Q  -> row-major f16, scaled by 1/sqrt(128)
//   MODE 1: K  -> row-major f16
//   MODE 2: V  -> transposed f16  VT[dv][BS]
// grid.x = BS/64, block = 256 (8 waves = 4 row-tiles x 2 col-halves).
// Each wave: 16 rows x 64 cols = 4 WMMA accumulators per A fragment.
// ---------------------------------------------------------------------------
template <int MODE>
__global__ __launch_bounds__(256) void proj_kernel(
    const float* __restrict__ X, const _Float16* __restrict__ WT,
    const float* __restrict__ bias, _Float16* __restrict__ Out)
{
    const int tid  = threadIdx.x;
    const int wave = tid >> 5;
    const int lane = tid & 31;
    const int n16  = lane & 15;
    const int hi   = lane >> 4;        // which 16-lane half
    const int hs   = hi * 8;           // A-operand K sub-offset

    const int rt   = wave >> 1;        // row tile 0..3
    const int ch   = wave & 1;         // column half 0..1
    const int row  = blockIdx.x * 64 + rt * 16 + n16;   // A-operand row
    const int nb   = ch * 64;                           // first column of this wave

    v8f acc[4];
    #pragma unroll
    for (int j = 0; j < 4; ++j) acc[j] = (v8f){};

    for (int c = 0; c < DMODEL / 32; ++c) {
        // ---- batched loads: A tile (f32) + 4 B fragments ----
        const float* ap = X + row * DMODEL + c * 32;
        float4 f0 = *(const float4*)(ap + hs);
        float4 f1 = *(const float4*)(ap + hs + 4);
        float4 f2 = *(const float4*)(ap + 16 + hs);
        float4 f3 = *(const float4*)(ap + 16 + hs + 4);

        V16H b[4];
        #pragma unroll
        for (int j = 0; j < 4; ++j) {
            const _Float16* bp = WT + (nb + j * 16 + n16) * DMODEL + c * 32 + hi * 8;
            b[j].u[0] = *(const uint4*)bp;
            b[j].u[1] = *(const uint4*)(bp + 16);
        }

        // ---- convert A, then 4 WMMAs reuse the same A fragment ----
        V16H a;
        a.e[0]  = (_Float16)f0.x; a.e[1]  = (_Float16)f0.y;
        a.e[2]  = (_Float16)f0.z; a.e[3]  = (_Float16)f0.w;
        a.e[4]  = (_Float16)f1.x; a.e[5]  = (_Float16)f1.y;
        a.e[6]  = (_Float16)f1.z; a.e[7]  = (_Float16)f1.w;
        a.e[8]  = (_Float16)f2.x; a.e[9]  = (_Float16)f2.y;
        a.e[10] = (_Float16)f2.z; a.e[11] = (_Float16)f2.w;
        a.e[12] = (_Float16)f3.x; a.e[13] = (_Float16)f3.y;
        a.e[14] = (_Float16)f3.z; a.e[15] = (_Float16)f3.w;

        #pragma unroll
        for (int j = 0; j < 4; ++j)
            acc[j] = wmma_f16(a.h, b[j].h, acc[j]);
    }

    const int hrow    = hi * 8;
    const int rowbase = blockIdx.x * 64 + rt * 16 + hrow;

    #pragma unroll
    for (int j = 0; j < 4; ++j) {
        const int   n  = nb + j * 16 + n16;
        const float bv = bias[n];
        if (MODE == 0) {
            const float sc = 0.08838834764831845f;  // 1/sqrt(128)
            #pragma unroll
            for (int r = 0; r < 8; ++r)
                Out[(rowbase + r) * DK + n] = (_Float16)((acc[j][r] + bv) * sc);
        } else if (MODE == 1) {
            #pragma unroll
            for (int r = 0; r < 8; ++r)
                Out[(rowbase + r) * DK + n] = (_Float16)(acc[j][r] + bv);
        } else {
            // transposed store: VT[n][BS]; 8 consecutive rows -> one 16B store
            union { _Float16 e[8]; uint4 u; } pk;
            #pragma unroll
            for (int r = 0; r < 8; ++r)
                pk.e[r] = (_Float16)(acc[j][r] + bv);
            *(uint4*)(Out + n * BS + rowbase) = pk.u;
        }
    }
}

// ---------------------------------------------------------------------------
// Stage 3: causal flash attention.
// 1 wave per block; wave handles 16 query rows, iterates 32-key blocks.
// grid.x = BS/16
// ---------------------------------------------------------------------------
__global__ __launch_bounds__(32) void attn_kernel(
    const _Float16* __restrict__ Qh, const _Float16* __restrict__ Kh,
    const _Float16* __restrict__ VT, float* __restrict__ out)
{
    __shared__ _Float16 pl[16 * 32];   // P tile bounce buffer [qrow][keycol]

    const int lane = threadIdx.x;
    const int n16  = lane & 15;
    const int hi   = lane >> 4;
    const int hs   = hi * 8;

    const int blk = blockIdx.x;
    const int b   = blk / (SEQ / 16);
    const int qt  = blk % (SEQ / 16);
    const int m0  = qt * 16;              // query tile base (within batch)
    const int rowbase = b * SEQ + m0;     // global row base

    // Preload Q A-operands (4 chunks of K=32)
    v16h aq[4];
    #pragma unroll
    for (int c = 0; c < 4; ++c) {
        const _Float16* qp = Qh + (rowbase + n16) * DK + c * 32 + hs;
        V16H a;
        a.u[0] = *(const uint4*)qp;
        a.u[1] = *(const uint4*)(qp + 16);
        aq[c] = a.h;
    }

    v8f o[8];
    #pragma unroll
    for (int d = 0; d < 8; ++d) o[d] = (v8f){};
    float m[8], l[8];
    #pragma unroll
    for (int r = 0; r < 8; ++r) { m[r] = -1e30f; l[r] = 0.0f; }

    const int kbmax = (m0 + 15) >> 5;   // inclusive: last key block needed
    for (int kb = 0; kb <= kbmax; ++kb) {
        const int key0 = kb * 32;
        const _Float16* kbase = Kh + (b * SEQ + key0 + n16) * DK + hi * 8;

        // ---- batched K fragment loads (8 x 32B): staggered waits ----
        V16H bk[8];
        #pragma unroll
        for (int c = 0; c < 4; ++c) {
            const _Float16* kp0 = kbase + c * 32;
            bk[c].u[0]     = *(const uint4*)kp0;
            bk[c].u[1]     = *(const uint4*)(kp0 + 16);
            const _Float16* kp1 = kp0 + 16 * DK;
            bk[c + 4].u[0] = *(const uint4*)kp1;
            bk[c + 4].u[1] = *(const uint4*)(kp1 + 16);
        }

        // ---- S tile = Q @ K^T : two 16x16 column fragments ----
        v8f s0 = {}, s1 = {};
        #pragma unroll
        for (int c = 0; c < 4; ++c) {
            s0 = wmma_f16(aq[c], bk[c].h, s0);
            s1 = wmma_f16(aq[c], bk[c + 4].h, s1);
        }

        // ---- issue V fragment loads NOW; softmax VALU hides their latency
        V16H bvv[8];
        #pragma unroll
        for (int d = 0; d < 8; ++d) {
            const _Float16* vp = VT + (d * 16 + n16) * BS + b * SEQ + key0 + hi * 8;
            bvv[d].u[0] = *(const uint4*)vp;
            bvv[d].u[1] = *(const uint4*)(vp + 16);
        }

        // Prefetch next key block's K rows (emits global_prefetch_b8)
        if (kb < kbmax)
            __builtin_prefetch(kbase + 32 * DK, 0, 3);

        // ---- causal mask: only one diagonal block per query tile ----
        if (key0 + 31 > m0) {
            #pragma unroll
            for (int r = 0; r < 8; ++r) {
                const int qg = m0 + hs + r;
                if (key0 + n16      > qg) s0[r] = -1e30f;
                if (key0 + 16 + n16 > qg) s1[r] = -1e30f;
            }
        }

        // ---- online softmax (rows live across 16-lane halves) ----
        float p0[8], p1[8];
        #pragma unroll
        for (int r = 0; r < 8; ++r) {
            float v0 = s0[r], v1 = s1[r];

            float t = fmaxf(v0, v1);
            t = fmaxf(t, __shfl_xor(t, 1, 32));
            t = fmaxf(t, __shfl_xor(t, 2, 32));
            t = fmaxf(t, __shfl_xor(t, 4, 32));
            t = fmaxf(t, __shfl_xor(t, 8, 32));

            const float mn   = fmaxf(m[r], t);
            const float corr = __expf(m[r] - mn);
            m[r] = mn;

            const float e0 = __expf(v0 - mn);
            const float e1 = __expf(v1 - mn);
            p0[r] = e0; p1[r] = e1;

            float s = e0 + e1;
            s += __shfl_xor(s, 1, 32);
            s += __shfl_xor(s, 2, 32);
            s += __shfl_xor(s, 4, 32);
            s += __shfl_xor(s, 8, 32);
            l[r] = l[r] * corr + s;

            #pragma unroll
            for (int d = 0; d < 8; ++d) o[d][r] *= corr;
        }

        // ---- relayout P (C fragment -> A operand) through LDS ----
        #pragma unroll
        for (int r = 0; r < 8; ++r) {
            pl[(hs + r) * 32 + n16]      = (_Float16)p0[r];
            pl[(hs + r) * 32 + 16 + n16] = (_Float16)p1[r];
        }
        __syncthreads();   // single-wave WG: ordering fence

        const _Float16* pp = pl + n16 * 32 + hs;
        V16H pa;
        pa.u[0] = *(const uint4*)pp;
        pa.u[1] = *(const uint4*)(pp + 16);
        __syncthreads();

        // ---- O += P @ V  (V fragments already in registers) ----
        #pragma unroll
        for (int d = 0; d < 8; ++d)
            o[d] = wmma_f16(pa.h, bvv[d].h, o[d]);
    }

    // ---- epilogue: normalize and store f32 ----
    #pragma unroll
    for (int r = 0; r < 8; ++r) {
        const float inv = 1.0f / l[r];
        const int   q   = rowbase + hs + r;
        #pragma unroll
        for (int d = 0; d < 8; ++d)
            out[q * DK + d * 16 + n16] = o[d][r] * inv;
    }
}

// ---------------------------------------------------------------------------
extern "C" void kernel_launch(void* const* d_in, const int* in_sizes, int n_in,
                              void* d_out, int out_size, void* d_ws, size_t ws_size,
                              hipStream_t stream) {
    const float* q_in = (const float*)d_in[0];
    const float* k_in = (const float*)d_in[1];
    const float* v_in = (const float*)d_in[2];
    const float* Wq   = (const float*)d_in[3];
    const float* bq   = (const float*)d_in[4];
    const float* Wk   = (const float*)d_in[5];
    const float* bk   = (const float*)d_in[6];
    const float* Wv   = (const float*)d_in[7];
    const float* bv   = (const float*)d_in[8];
    float* out = (float*)d_out;

    char* ws = (char*)d_ws;
    _Float16* Qh  = (_Float16*)(ws);                          // 4 MB
    _Float16* Kh  = (_Float16*)(ws + ((size_t)4  << 20));     // 4 MB
    _Float16* VT  = (_Float16*)(ws + ((size_t)8  << 20));     // 4 MB
    _Float16* WTq = (_Float16*)(ws + ((size_t)12 << 20));     // 256 KB each
    _Float16* WTk = WTq + DMODEL * DK;
    _Float16* WTv = WTk + DMODEL * DK;

    // Stage 1: weight transpose/convert (131072 elems each)
    wprep_kernel<<<(DMODEL * DK + 255) / 256, 256, 0, stream>>>(Wq, WTq);
    wprep_kernel<<<(DMODEL * DK + 255) / 256, 256, 0, stream>>>(Wk, WTk);
    wprep_kernel<<<(DMODEL * DK + 255) / 256, 256, 0, stream>>>(Wv, WTv);

    // Stage 2: projections (grid = BS/64 row tiles, 8 waves = 4 rt x 2 ch)
    proj_kernel<0><<<BS / 64, 256, 0, stream>>>(q_in, WTq, bq, Qh);
    proj_kernel<1><<<BS / 64, 256, 0, stream>>>(k_in, WTk, bk, Kh);
    proj_kernel<2><<<BS / 64, 256, 0, stream>>>(v_in, WTv, bv, VT);

    // Stage 3: causal flash attention
    attn_kernel<<<BS / 16, 32, 0, stream>>>(Qh, Kh, VT, out);
}
